// Model_39676907883593
// MI455X (gfx1250) — matrix operand order repaired and checked by reading, not compiled, not run on hardware
//
#include <hip/hip_runtime.h>
#include <hip/hip_bf16.h>

typedef __attribute__((ext_vector_type(16))) _Float16 v16h;
typedef __attribute__((ext_vector_type(8)))  _Float16 v8h;
typedef __attribute__((ext_vector_type(8)))  float    v8f;

#define S_LEN 2048
#define D_QK  64
#define D_V   512
#define BQ    32
#define BK    64
#define SOFT_SCALE (1.0f/64.0f)
#define KEEP_P 0.75f

// ---- helpers -------------------------------------------------------------

__device__ __forceinline__ v16h ld2(const _Float16* p0, const _Float16* p1) {
    union { v16h v; v8h h[2]; } u;
    u.h[0] = *(const v8h*)p0;
    u.h[1] = *(const v8h*)p1;
    return u.v;
}

__device__ __forceinline__ unsigned hash_u32(unsigned x) {
    x *= 0x9E3779B9u; x ^= x >> 16;
    x *= 0x7FEB352Du; x ^= x >> 15;
    x *= 0x846CA68Bu; x ^= x >> 16;
    return x;
}

// ---- prep kernel 1: K_f16[s][d] = sum_c n[s][c]*m[d][c] ------------------

__global__ __launch_bounds__(256) void build_k_f16(
    const float* __restrict__ mm, const float* __restrict__ nn,
    _Float16* __restrict__ kf)
{
    int t = blockIdx.x * 256 + threadIdx.x;      // S_LEN*D_QK total
    int s = t >> 6, d = t & 63;
    const float4* nr = (const float4*)(nn + (size_t)s * 512);
    const float4* mr = (const float4*)(mm + (size_t)d * 512);
    float acc = 0.0f;
    #pragma unroll 4
    for (int c = 0; c < 128; ++c) {
        float4 a = nr[c], b = mr[c];
        acc += a.x * b.x + a.y * b.y + a.z * b.z + a.w * b.w;
    }
    kf[t] = (_Float16)acc;
}

// ---- prep kernel 2: Vt_f16[dv][s] = (half)q[s][dv] (tiled transpose) -----

__global__ __launch_bounds__(256) void build_vt_f16(
    const float* __restrict__ q, _Float16* __restrict__ vt)
{
    __shared__ float tile[32][33];
    const int tilesPerRow = D_V / 32;            // 16
    int bx = blockIdx.x % tilesPerRow;           // dv tile
    int by = blockIdx.x / tilesPerRow;           // s tile
    int tid = threadIdx.x;
    #pragma unroll
    for (int i = 0; i < 4; ++i) {
        int e = tid + i * 256; int r = e >> 5, c = e & 31;
        tile[r][c] = q[(size_t)(by * 32 + r) * D_V + bx * 32 + c];
    }
    __syncthreads();
    #pragma unroll
    for (int i = 0; i < 4; ++i) {
        int e = tid + i * 256; int r = e >> 5, c = e & 31;
        vt[(size_t)(bx * 32 + r) * S_LEN + by * 32 + c] = (_Float16)tile[c][r];
    }
}

// ---- fused flash-attention kernel ----------------------------------------
// grid: 16 batches * (S/BQ) query tiles; block: 256 threads = 8 waves.
// wave(w): score tile  rows m0=(w>>2)*16, keys n0=(w&3)*16
//          PV tile     rows m0,          dv  dv0=(w&3)*128 (8 n-tiles of 16)

__global__ __launch_bounds__(256) void fa_fused(
    const float* __restrict__ x1, const _Float16* __restrict__ kf,
    const _Float16* __restrict__ vt, float* __restrict__ out)
{
    __shared__ float    sc[BQ * BK];      // raw f32 scores
    __shared__ _Float16 pshm[BQ * BK];    // probs (post-dropout) f16
    __shared__ float    rescSh[BQ];
    __shared__ float    linvSh[BQ];

    const int tid    = threadIdx.x;
    const int lane   = tid & 31;
    const int w      = tid >> 5;
    const int laneLo = lane & 15;
    const int laneHi = lane >> 4;

    const int b     = blockIdx.x >> 6;
    const int qbase = (blockIdx.x & 63) * BQ;

    const int m0  = (w >> 2) * 16;
    const int n0s = (w & 3) * 16;
    const int dv0 = (w & 3) * 128;

    // Q A-fragments (kept in registers for the whole kernel), converted f32->f16.
    // A layout (16-bit, 16x32): halfs 0..7 -> K = kk + laneHi*8 + h,
    //                           halfs 8..15 -> K = kk + 16 + laneHi*8 + (h-8).
    v16h aq0, aq1;
    {
        const float* qrow = x1 + (size_t)(b * S_LEN + qbase + m0 + laneLo) * D_QK;
        const int koff = laneHi * 8;
        union { v16h v; _Float16 e[16]; } u0, u1;
        #pragma unroll
        for (int h = 0; h < 8; ++h) {
            u0.e[h]     = (_Float16)qrow[koff + h];
            u0.e[h + 8] = (_Float16)qrow[koff + 16 + h];
            u1.e[h]     = (_Float16)qrow[32 + koff + h];
            u1.e[h + 8] = (_Float16)qrow[32 + koff + 16 + h];
        }
        aq0 = u0.v; aq1 = u1.v;
    }

    v8f acc[8];
    #pragma unroll
    for (int i = 0; i < 8; ++i) acc[i] = (v8f){};

    // online-softmax state: row = tid>>3, replicated across 8 lanes of its group
    const int srow = tid >> 3;
    const int ssub = tid & 7;
    float m_i = -1e30f;
    float l_i = 0.0f;
    const unsigned rowg = (unsigned)(b * S_LEN + qbase + srow);

    for (int j = 0; j < S_LEN; j += BK) {
        // ---- scores: 16x16 tile per wave, K=64 in two WMMAs ----
        {
            const _Float16* kr = kf + (size_t)(j + n0s + laneLo) * D_QK + laneHi * 8;
            v16h b0 = ld2(kr,      kr + 16);   // d = laneHi*8 + {0..7, 16..23}
            v16h b1 = ld2(kr + 32, kr + 48);   // d = 32 + laneHi*8 + {0..7, 16..23}
            v8f c = (v8f){};
            c = __builtin_amdgcn_wmma_f32_16x16x32_f16(false, aq0, false, b0,
                                                       (short)0, c, false, false);
            c = __builtin_amdgcn_wmma_f32_16x16x32_f16(false, aq1, false, b1,
                                                       (short)0, c, false, false);
            // C layout: VGPR r -> row m0 + laneHi*8 + r, col n0s + laneLo
            float* dst = sc + (m0 + laneHi * 8) * BK + n0s + laneLo;
            #pragma unroll
            for (int r = 0; r < 8; ++r) dst[r * BK] = c[r];
            // prefetch next K tile into caches
            if (j + BK < S_LEN)
                __builtin_prefetch(kf + (size_t)(j + BK + n0s + laneLo) * D_QK, 0, 3);
        }
        __syncthreads();

        // ---- online softmax + dropout (8 lanes per row) ----
        {
            const float* srcp = sc + srow * BK + ssub * 8;
            float v[8];
            #pragma unroll
            for (int i = 0; i < 8; ++i) v[i] = srcp[i] * SOFT_SCALE;
            float mx = v[0];
            #pragma unroll
            for (int i = 1; i < 8; ++i) mx = fmaxf(mx, v[i]);
            #pragma unroll
            for (int msk = 1; msk < 8; msk <<= 1)
                mx = fmaxf(mx, __shfl_xor(mx, msk, 32));
            float mnew = fmaxf(m_i, mx);
            float resc = __expf(m_i - mnew);
            float sum = 0.0f;
            union { v8h v; _Float16 e[8]; } pu;
            #pragma unroll
            for (int i = 0; i < 8; ++i) {
                float e = __expf(v[i] - mnew);
                sum += e;
                unsigned idx = rowg * (unsigned)S_LEN + (unsigned)(j + ssub * 8 + i);
                unsigned h = hash_u32(idx);
                float p = ((h >> 8) < 12582912u) ? e * (1.0f / KEEP_P) : 0.0f;
                pu.e[i] = (_Float16)p;
            }
            *(v8h*)(pshm + srow * BK + ssub * 8) = pu.v;
            #pragma unroll
            for (int msk = 1; msk < 8; msk <<= 1)
                sum += __shfl_xor(sum, msk, 32);
            l_i = l_i * resc + sum;
            m_i = mnew;
            if (ssub == 0) rescSh[srow] = resc;
        }
        __syncthreads();

        // ---- rescale accumulators + P @ V (16 WMMAs per wave) ----
        {
            const float* rsp = rescSh + m0 + laneHi * 8;
            float rs[8];
            #pragma unroll
            for (int r = 0; r < 8; ++r) rs[r] = rsp[r];
            #pragma unroll
            for (int nt = 0; nt < 8; ++nt)
                #pragma unroll
                for (int r = 0; r < 8; ++r) acc[nt][r] *= rs[r];

            const _Float16* pr = pshm + (m0 + laneLo) * BK + laneHi * 8;
            v16h ap0 = ld2(pr,      pr + 16);  // K = laneHi*8 + {0..7, 16..23}
            v16h ap1 = ld2(pr + 32, pr + 48);  // K = 32 + ...

            const _Float16* vbase = vt + (size_t)(dv0 + laneLo) * S_LEN + j + laneHi * 8;
            #pragma unroll
            for (int nt = 0; nt < 8; ++nt) {
                const _Float16* vr = vbase + (size_t)nt * 16 * S_LEN;
                v16h b0 = ld2(vr,      vr + 16);   // keys laneHi*8 + {0..7, 16..23}
                v16h b1 = ld2(vr + 32, vr + 48);   // keys 32 + laneHi*8 + {0..7, 16..23}
                acc[nt] = __builtin_amdgcn_wmma_f32_16x16x32_f16(false, ap0, false, b0,
                                                                 (short)0, acc[nt], false, false);
                acc[nt] = __builtin_amdgcn_wmma_f32_16x16x32_f16(false, ap1, false, b1,
                                                                 (short)0, acc[nt], false, false);
            }
        }
        __syncthreads();
    }

    // ---- epilogue: divide by softmax sum, store ----
    if (ssub == 0) linvSh[srow] = 1.0f / l_i;
    __syncthreads();
    {
        const float* lp = linvSh + m0 + laneHi * 8;
        float li[8];
        #pragma unroll
        for (int r = 0; r < 8; ++r) li[r] = lp[r];
        float* ob = out + (size_t)(b * S_LEN + qbase + m0 + laneHi * 8) * D_V
                        + dv0 + laneLo;
        #pragma unroll
        for (int nt = 0; nt < 8; ++nt)
            #pragma unroll
            for (int r = 0; r < 8; ++r)
                ob[(size_t)r * D_V + nt * 16] = acc[nt][r] * li[r];
    }
}

// ---- host launch ---------------------------------------------------------

extern "C" void kernel_launch(void* const* d_in, const int* in_sizes, int n_in,
                              void* d_out, int out_size, void* d_ws, size_t ws_size,
                              hipStream_t stream) {
    const float* x1 = (const float*)d_in[0];   // [16, 2048, 64]
    const float* m  = (const float*)d_in[1];   // [64, 512]
    const float* n  = (const float*)d_in[2];   // [2048, 512]
    const float* q  = (const float*)d_in[3];   // [2048, 512]
    float* out = (float*)d_out;                // [16, 2048, 512]

    _Float16* kf = (_Float16*)d_ws;                              // 256 KB
    _Float16* vt = (_Float16*)((char*)d_ws + 512 * 1024);        // 2 MB

    build_k_f16<<<(S_LEN * D_QK) / 256, 256, 0, stream>>>(m, n, kf);
    build_vt_f16<<<(S_LEN / 32) * (D_V / 32), 256, 0, stream>>>(q, vt);
    fa_fused<<<16 * (S_LEN / BQ), 256, 0, stream>>>(x1, kf, vt, out);
}
